// MAB_24601572672032
// MI455X (gfx1250) — matrix operand order repaired and checked by reading, not compiled, not run on hardware
//
#include <hip/hip_runtime.h>
#include <hip/hip_bf16.h>

typedef __attribute__((ext_vector_type(16))) _Float16 v16h;
typedef __attribute__((ext_vector_type(8)))  _Float16 v8h;
typedef __attribute__((ext_vector_type(8)))  float    v8f;
typedef __attribute__((ext_vector_type(4)))  float    v4f;

#define S_LEN 1024
#define BATCH 4
#define EMB   1024
#define HEADS 16
#define DHEAD 64
#define DFF_  4096
#define NTOK  4096   // S*B tokens

static __device__ __forceinline__ v16h cat8(v8h lo, v8h hi) {
  return __builtin_shufflevector(lo, hi, 0,1,2,3,4,5,6,7,8,9,10,11,12,13,14,15);
}
static __device__ __forceinline__ v8f zero8() {
  v8f z = {0.f,0.f,0.f,0.f,0.f,0.f,0.f,0.f};
  return z;
}
static __device__ __forceinline__ v8f wmma_f16(v16h a, v16h b, v8f c) {
  return __builtin_amdgcn_wmma_f32_16x16x32_f16(false, a, false, b, (short)0, c, false, false);
}

// ---------------------------------------------------------------- fp32 -> f16
__global__ __launch_bounds__(256)
void cvt_f32_to_f16(const float* __restrict__ in, _Float16* __restrict__ out, int n) {
  int i = blockIdx.x * blockDim.x + threadIdx.x;
  if (i < n) out[i] = (_Float16)in[i];
}

// ---------------------------------------------------------------- LayerNorm (one token row per block)
__global__ __launch_bounds__(256)
void ln_kernel(const float* __restrict__ x, const float* __restrict__ g,
               const float* __restrict__ b, _Float16* __restrict__ out) {
  __shared__ float sred[18];
  int row = blockIdx.x;
  int tid = threadIdx.x;
  const float* xr = x + (size_t)row * EMB;
  v4f v = *(const v4f*)(xr + tid * 4);
  float s = v[0] + v[1] + v[2] + v[3];
  float q = v[0]*v[0] + v[1]*v[1] + v[2]*v[2] + v[3]*v[3];
  #pragma unroll
  for (int off = 16; off >= 1; off >>= 1) {
    s += __shfl_xor(s, off, 32);
    q += __shfl_xor(q, off, 32);
  }
  int wid = tid >> 5, lane = tid & 31;
  if (lane == 0) { sred[wid] = s; sred[8 + wid] = q; }
  __syncthreads();
  if (tid == 0) {
    float ts = 0.f, tq = 0.f;
    for (int i = 0; i < 8; ++i) { ts += sred[i]; tq += sred[8 + i]; }
    float mu  = ts * (1.0f / EMB);
    float var = tq * (1.0f / EMB) - mu * mu;
    sred[16] = mu;
    sred[17] = rsqrtf(var + 1e-5f);
  }
  __syncthreads();
  float mu = sred[16], rs = sred[17];
  #pragma unroll
  for (int j = 0; j < 4; ++j) {
    int c = tid * 4 + j;
    float val = (v[j] - mu) * rs * g[c] + b[c];
    out[(size_t)row * EMB + c] = (_Float16)val;
  }
}

// ---------------------------------------------------------------- generic WMMA GEMM
// C[M,N] = A[M,K] (f16) x W[N,K]^T (f16) + bias;  optional exact GELU; writes
// fp32 (+residual) and/or f16 copies.
// Block tile 128(M) x 128(N), 8 waves; each wave: 32(M) x 64(N) =
// 2 A-frags x 4 B-frags = 8 WMMAs per 32-wide K-step (1.5 ds_load_b128/WMMA).
__global__ __launch_bounds__(256)
void gemm_kernel(const _Float16* __restrict__ A, const _Float16* __restrict__ W,
                 const float* __restrict__ bias, float* __restrict__ outf,
                 const float* __restrict__ resid, _Float16* __restrict__ outh,
                 int M, int N, int K, int act)
{
  __shared__ __align__(16) _Float16 As[128 * 40];
  __shared__ __align__(16) _Float16 Bs[128 * 40];
  int tid  = threadIdx.x;
  int lane = tid & 31;
  int w    = tid >> 5;       // 0..7
  int mw   = w >> 1;         // 0..3  (32-row band)
  int nw   = w & 1;          // 0..1  (64-col band)
  int m0   = blockIdx.y * 128;
  int n0   = blockIdx.x * 128;

  v8f acc[2][4];
  #pragma unroll
  for (int mi = 0; mi < 2; ++mi)
    #pragma unroll
    for (int i = 0; i < 4; ++i) acc[mi][i] = zero8();

  int tr = tid >> 1;           // 0..127 (tile row for cooperative loads)
  int tk = (tid & 1) * 16;     // k-segment (two v8h each)

  for (int k0 = 0; k0 < K; k0 += 32) {
    __syncthreads();
    *(v8h*)(&As[tr * 40 + tk])     = *(const v8h*)(&A[(size_t)(m0 + tr) * K + k0 + tk]);
    *(v8h*)(&As[tr * 40 + tk + 8]) = *(const v8h*)(&A[(size_t)(m0 + tr) * K + k0 + tk + 8]);
    *(v8h*)(&Bs[tr * 40 + tk])     = *(const v8h*)(&W[(size_t)(n0 + tr) * K + k0 + tk]);
    *(v8h*)(&Bs[tr * 40 + tk + 8]) = *(const v8h*)(&W[(size_t)(n0 + tr) * K + k0 + tk + 8]);
    if (k0 + 32 < K) {  // pull next tiles toward the WGP (global_prefetch_b8)
      __builtin_prefetch(&A[(size_t)(m0 + tr) * K + k0 + 32 + tk], 0, 1);
      __builtin_prefetch(&W[(size_t)(n0 + tr) * K + k0 + 32 + tk], 0, 1);
    }
    __syncthreads();

    // B fragments first (shared by both A bands): N = lane&15, kbase = (lane>>4)*8
    int kb2 = (lane >> 4) * 8;
    v16h bfrag[4];
    #pragma unroll
    for (int nd = 0; nd < 4; ++nd) {
      int bn = nw * 64 + nd * 16 + (lane & 15);
      bfrag[nd] = cat8(*(const v8h*)(&Bs[bn * 40 + kb2]),
                       *(const v8h*)(&Bs[bn * 40 + kb2 + 16]));
    }
    // Two A fragments: 16-bit A layout, M = lane&15, kbase = (lane>>4)*8
    int kb = (lane >> 4) * 8;
    #pragma unroll
    for (int mi = 0; mi < 2; ++mi) {
      int am = mw * 32 + mi * 16 + (lane & 15);
      v16h afrag = cat8(*(const v8h*)(&As[am * 40 + kb]),
                        *(const v8h*)(&As[am * 40 + kb + 16]));
      #pragma unroll
      for (int nd = 0; nd < 4; ++nd)
        acc[mi][nd] = wmma_f16(afrag, bfrag[nd], acc[mi][nd]);
    }
  }

  int row0 = (lane >> 4) * 8;
  int col  = lane & 15;
  #pragma unroll
  for (int mi = 0; mi < 2; ++mi) {
    #pragma unroll
    for (int nd = 0; nd < 4; ++nd) {
      int gn = n0 + nw * 64 + nd * 16 + col;
      float bval = bias[gn];
      #pragma unroll
      for (int v = 0; v < 8; ++v) {
        int gm = m0 + mw * 32 + mi * 16 + row0 + v;
        float val = acc[mi][nd][v] + bval;
        if (act == 1) val = 0.5f * val * (1.0f + erff(val * 0.70710678118654752f));
        size_t idx = (size_t)gm * N + gn;
        if (outf) outf[idx] = val + (resid ? resid[idx] : 0.0f);
        if (outh) outh[idx] = (_Float16)val;
      }
    }
  }
}

// ---------------------------------------------------------------- RoPE + head reshape
// qkv_h[token][3E] -> Qr/Kr/Vr [B*H][S][D] f16; Q pre-scaled by 1/sqrt(D)
__global__ __launch_bounds__(256)
void rope_kernel(const _Float16* __restrict__ qkv,
                 const float* __restrict__ cosb, const float* __restrict__ sinb,
                 _Float16* __restrict__ Qr, _Float16* __restrict__ Kr,
                 _Float16* __restrict__ Vr)
{
  int idx = blockIdx.x * blockDim.x + threadIdx.x;  // S*B*H*32
  int i = idx & 31;
  int h = (idx >> 5) & (HEADS - 1);
  int b = (idx >> 9) & (BATCH - 1);
  int s = idx >> 11;
  size_t tok = (size_t)s * BATCH + b;
  const _Float16* rowq = qkv + tok * (3 * EMB);
  float c  = cosb[s * 32 + i];
  float sn = sinb[s * 32 + i];
  int colh = h * DHEAD + i;
  float q1 = (float)rowq[colh],            q2 = (float)rowq[colh + 32];
  float k1 = (float)rowq[EMB + colh],      k2 = (float)rowq[EMB + colh + 32];
  float v1 = (float)rowq[2 * EMB + colh],  v2 = (float)rowq[2 * EMB + colh + 32];
  size_t obase = ((size_t)(b * HEADS + h) * S_LEN + s) * DHEAD;
  const float scale = 0.125f; // 1/sqrt(64)
  Qr[obase + i]      = (_Float16)((q1 * c - q2 * sn) * scale);
  Qr[obase + 32 + i] = (_Float16)((q1 * sn + q2 * c) * scale);
  Kr[obase + i]      = (_Float16)(k1 * c - k2 * sn);
  Kr[obase + 32 + i] = (_Float16)(k1 * sn + k2 * c);
  Vr[obase + i]      = (_Float16)v1;
  Vr[obase + 32 + i] = (_Float16)v2;
}

// ---------------------------------------------------------------- flash attention
// grid (S/64, B*H), 128 threads; wave w handles 16 query rows.
__global__ __launch_bounds__(128)
void attn_kernel(const _Float16* __restrict__ Qr, const _Float16* __restrict__ Kr,
                 const _Float16* __restrict__ Vr, _Float16* __restrict__ aoh)
{
  __shared__ __align__(16) _Float16 Ks[32 * 72];      // K tile  [t][d]
  __shared__ __align__(16) _Float16 Vs[64 * 40];      // V tile transposed [d][t]
  __shared__ __align__(16) _Float16 Pw[4 * 16 * 40];  // per-wave P staging

  int tid   = threadIdx.x;
  int lane  = tid & 31;
  int w     = tid >> 5;                  // 0..3
  int bh    = blockIdx.y;                // 0..B*H-1
  int qbase = blockIdx.x * 64 + w * 16;

  const _Float16* Qb = Qr + (size_t)bh * S_LEN * DHEAD;
  const _Float16* Kb = Kr + (size_t)bh * S_LEN * DHEAD;
  const _Float16* Vb = Vr + (size_t)bh * S_LEN * DHEAD;

  int m  = lane & 15;
  int kb = (lane >> 4) * 8;
  const _Float16* qrow = Qb + (size_t)(qbase + m) * DHEAD;
  v16h aq0 = cat8(*(const v8h*)(qrow + kb),      *(const v8h*)(qrow + kb + 16));
  v16h aq1 = cat8(*(const v8h*)(qrow + 32 + kb), *(const v8h*)(qrow + 48 + kb));

  v8f o[4];
  #pragma unroll
  for (int i = 0; i < 4; ++i) o[i] = zero8();
  float rmax[8], rsum[8];
  #pragma unroll
  for (int v = 0; v < 8; ++v) { rmax[v] = -1e30f; rsum[v] = 0.0f; }

  _Float16* pw = &Pw[w * 16 * 40];

  for (int kt = 0; kt < S_LEN; kt += 32) {
    __syncthreads();   // protects Ks/Vs reuse from previous iteration
    {
      int r = tid >> 2;                  // 0..31 (key row in tile)
      int dseg = (tid & 3) * 16;
      const v8h* ksrc = (const v8h*)(Kb + (size_t)(kt + r) * DHEAD + dseg);
      *(v8h*)(&Ks[r * 72 + dseg])     = ksrc[0];
      *(v8h*)(&Ks[r * 72 + dseg + 8]) = ksrc[1];
      const _Float16* vsrc = Vb + (size_t)(kt + r) * DHEAD + dseg;
      #pragma unroll
      for (int j = 0; j < 16; ++j) Vs[(dseg + j) * 40 + r] = vsrc[j];
    }
    __syncthreads();

    // scores: 16 q-rows x 32 keys = two C fragments
    v8f sf[2];
    #pragma unroll
    for (int tg = 0; tg < 2; ++tg) {
      int tl = (lane & 15) + tg * 16;
      int dk = (lane >> 4) * 8;
      v16h b0 = cat8(*(const v8h*)(&Ks[tl * 72 + dk]),
                     *(const v8h*)(&Ks[tl * 72 + dk + 16]));
      v16h b1 = cat8(*(const v8h*)(&Ks[tl * 72 + 32 + dk]),
                     *(const v8h*)(&Ks[tl * 72 + 32 + dk + 16]));
      v8f c = zero8();
      c = wmma_f16(aq0, b0, c);
      c = wmma_f16(aq1, b1, c);
      sf[tg] = c;
    }

    // online softmax (row lives across a 16-lane half; reduce with shfl_xor w=16)
    #pragma unroll
    for (int v = 0; v < 8; ++v) {
      float mloc = fmaxf(sf[0][v], sf[1][v]);
      #pragma unroll
      for (int off = 8; off >= 1; off >>= 1) mloc = fmaxf(mloc, __shfl_xor(mloc, off, 16));
      float mnew  = fmaxf(rmax[v], mloc);
      float alpha = __expf(rmax[v] - mnew);
      float p0 = __expf(sf[0][v] - mnew);
      float p1 = __expf(sf[1][v] - mnew);
      float ps = p0 + p1;
      #pragma unroll
      for (int off = 8; off >= 1; off >>= 1) ps += __shfl_xor(ps, off, 16);
      rsum[v] = rsum[v] * alpha + ps;
      rmax[v] = mnew;
      #pragma unroll
      for (int nd = 0; nd < 4; ++nd) o[nd][v] *= alpha;
      int prow = (lane >> 4) * 8 + v;
      pw[prow * 40 + (lane & 15)]      = (_Float16)p0;
      pw[prow * 40 + (lane & 15) + 16] = (_Float16)p1;
    }
    __syncthreads();

    // P x V  (contraction over 32 keys)
    v16h ap = cat8(*(const v8h*)(&pw[m * 40 + kb]),
                   *(const v8h*)(&pw[m * 40 + kb + 16]));
    int ktl = (lane >> 4) * 8;
    #pragma unroll
    for (int nd = 0; nd < 4; ++nd) {
      int dcol = nd * 16 + (lane & 15);
      v16h bv = cat8(*(const v8h*)(&Vs[dcol * 40 + ktl]),
                     *(const v8h*)(&Vs[dcol * 40 + ktl + 16]));
      o[nd] = wmma_f16(ap, bv, o[nd]);
    }
  }

  int b    = bh >> 4;      // bh / HEADS
  int h    = bh & (HEADS - 1);
  int row0 = (lane >> 4) * 8;
  int col  = lane & 15;
  #pragma unroll
  for (int v = 0; v < 8; ++v) {
    float inv = 1.0f / rsum[v];
    int srow = qbase + row0 + v;
    size_t tok = (size_t)srow * BATCH + b;
    #pragma unroll
    for (int nd = 0; nd < 4; ++nd) {
      aoh[tok * EMB + h * DHEAD + nd * 16 + col] = (_Float16)(o[nd][v] * inv);
    }
  }
}

// ----------------------------------------------------------------
extern "C" void kernel_launch(void* const* d_in, const int* in_sizes, int n_in,
                              void* d_out, int out_size, void* d_ws, size_t ws_size,
                              hipStream_t stream)
{
  const float* q        = (const float*)d_in[0];
  const float* in_w     = (const float*)d_in[1];
  const float* in_b     = (const float*)d_in[2];
  const float* out_w    = (const float*)d_in[3];
  const float* out_b    = (const float*)d_in[4];
  const float* ln1_g    = (const float*)d_in[5];
  const float* ln1_b    = (const float*)d_in[6];
  const float* ln2_g    = (const float*)d_in[7];
  const float* ln2_b    = (const float*)d_in[8];
  const float* ff1_w    = (const float*)d_in[9];
  const float* ff1_b    = (const float*)d_in[10];
  const float* ff2_w    = (const float*)d_in[11];
  const float* ff2_b    = (const float*)d_in[12];
  const float* rope_cos = (const float*)d_in[13];
  const float* rope_sin = (const float*)d_in[14];
  float* out = (float*)d_out;
  (void)in_sizes; (void)n_in; (void)out_size; (void)ws_size;

  char* ws = (char*)d_ws;
  size_t off = 0;
  auto alloc = [&](size_t bytes) -> char* {
    char* p = ws + off;
    off += (bytes + 255) & ~(size_t)255;
    return p;
  };

  _Float16* wh_qkv = (_Float16*)alloc((size_t)3 * EMB * EMB * 2);
  _Float16* wh_out = (_Float16*)alloc((size_t)EMB * EMB * 2);
  _Float16* wh_ff1 = (_Float16*)alloc((size_t)DFF_ * EMB * 2);
  _Float16* wh_ff2 = (_Float16*)alloc((size_t)EMB * DFF_ * 2);
  _Float16* xh     = (_Float16*)alloc((size_t)NTOK * EMB * 2);
  _Float16* qkv_h  = (_Float16*)alloc((size_t)NTOK * 3 * EMB * 2);
  _Float16* Qr     = (_Float16*)alloc((size_t)BATCH * HEADS * S_LEN * DHEAD * 2);
  _Float16* Kr     = (_Float16*)alloc((size_t)BATCH * HEADS * S_LEN * DHEAD * 2);
  _Float16* Vr     = (_Float16*)alloc((size_t)BATCH * HEADS * S_LEN * DHEAD * 2);
  _Float16* aoh    = (_Float16*)alloc((size_t)NTOK * EMB * 2);
  float*    x1     = (float*)   alloc((size_t)NTOK * EMB * 4);
  _Float16* yh     = (_Float16*)alloc((size_t)NTOK * EMB * 2);
  _Float16* ffh    = (_Float16*)alloc((size_t)NTOK * DFF_ * 2);

  // 1) weight casts to f16
  {
    int n;
    n = 3 * EMB * EMB; cvt_f32_to_f16<<<(n + 255) / 256, 256, 0, stream>>>(in_w, wh_qkv, n);
    n = EMB * EMB;     cvt_f32_to_f16<<<(n + 255) / 256, 256, 0, stream>>>(out_w, wh_out, n);
    n = DFF_ * EMB;    cvt_f32_to_f16<<<(n + 255) / 256, 256, 0, stream>>>(ff1_w, wh_ff1, n);
    n = EMB * DFF_;    cvt_f32_to_f16<<<(n + 255) / 256, 256, 0, stream>>>(ff2_w, wh_ff2, n);
  }

  // 2) LN1
  ln_kernel<<<NTOK, 256, 0, stream>>>(q, ln1_g, ln1_b, xh);

  // 3) QKV projection (4096 x 3072 x 1024)
  gemm_kernel<<<dim3(3 * EMB / 128, NTOK / 128), 256, 0, stream>>>(
      xh, wh_qkv, in_b, nullptr, nullptr, qkv_h, NTOK, 3 * EMB, EMB, 0);

  // 4) RoPE + head split
  rope_kernel<<<(S_LEN * BATCH * HEADS * 32) / 256, 256, 0, stream>>>(
      qkv_h, rope_cos, rope_sin, Qr, Kr, Vr);

  // 5) attention
  attn_kernel<<<dim3(S_LEN / 64, BATCH * HEADS), 128, 0, stream>>>(Qr, Kr, Vr, aoh);

  // 6) out projection + residual -> x1 (fp32)
  gemm_kernel<<<dim3(EMB / 128, NTOK / 128), 256, 0, stream>>>(
      aoh, wh_out, out_b, x1, q, nullptr, NTOK, EMB, EMB, 0);

  // 7) LN2
  ln_kernel<<<NTOK, 256, 0, stream>>>(x1, ln2_g, ln2_b, yh);

  // 8) FF1 + exact GELU (4096 x 4096 x 1024)
  gemm_kernel<<<dim3(DFF_ / 128, NTOK / 128), 256, 0, stream>>>(
      yh, wh_ff1, ff1_b, nullptr, nullptr, ffh, NTOK, DFF_, EMB, 1);

  // 9) FF2 + residual -> output (4096 x 1024 x 4096)
  gemm_kernel<<<dim3(EMB / 128, NTOK / 128), 256, 0, stream>>>(
      ffh, wh_ff2, ff2_b, out, x1, nullptr, NTOK, EMB, DFF_, 0);
}
